// CoSADUV_NoTemporal_57191784513757
// MI455X (gfx1250) — matrix operand order repaired and checked by reading, not compiled, not run on hardware
//
#include <hip/hip_runtime.h>
#include <math.h>

// ---------------------------------------------------------------------------
// CoSADUV_NoTemporal for MI455X (gfx1250): bf16 WMMA GEMMs + persistent
// wave32 LSTM scan kernels. Matrix math uses v_wmma_f32_16x16x32_bf16;
// GEMM tile staging uses global_load_async_to_lds_b128 (ASYNCcnt);
// the serial scan prefetches the next timestep's gx tile.
// ---------------------------------------------------------------------------

typedef __bf16 bf16_t;
typedef __attribute__((ext_vector_type(16))) __bf16 v16bf;
typedef __attribute__((ext_vector_type(8)))  float  v8f;

#define DEVINL __device__ __forceinline__

DEVINL float sigmf(float x) { return 1.0f / (1.0f + __expf(-x)); }

// LDS byte offset of a __shared__ object: generic (flat) address low 32 bits
// are the LDS offset (ISA aperture rule).
DEVINL unsigned lds_off(const void* p) { return (unsigned)(uintptr_t)p; }

// A-fragment (16x32 MxK, bf16): lane l holds row M=l&15; element e maps to
//   K = ((l>>4)<<4) + e
// B-fragment (32x16 KxN, bf16): lane l holds col N=l&15; element e maps to
//   K = ((l>>4)<<4) + e
// C/D f32 (16x16): element r maps to M = ((l>>4)<<3)+r, N = l&15.

// ---------------------------------------------------------------------------
// Scene-context FCs: sc1[8,512] = ctx@fc1_w^T + b ; scr[8,256] likewise.
// ---------------------------------------------------------------------------
__global__ void fc_scene(const float* __restrict__ ctx,
                         const float* __restrict__ w1, const float* __restrict__ b1,
                         const float* __restrict__ wr, const float* __restrict__ br,
                         float* __restrict__ sc1, float* __restrict__ scr) {
  int idx = blockIdx.x * blockDim.x + threadIdx.x;
  if (idx < 8 * 512) {
    int n = idx >> 9, o = idx & 511;
    float acc = b1[o];
    for (int k = 0; k < 128; ++k) acc += ctx[n * 128 + k] * w1[o * 128 + k];
    sc1[idx] = acc;
  } else if (idx < 8 * 512 + 8 * 256) {
    int r = idx - 8 * 512;
    int n = r >> 8, o = r & 255;
    float acc = br[o];
    for (int k = 0; k < 128; ++k) acc += ctx[n * 128 + k] * wr[o * 128 + k];
    scr[r] = acc;
  }
}

// ---------------------------------------------------------------------------
// f32 -> bf16 conversion (weights staging).
// ---------------------------------------------------------------------------
__global__ void to_bf16(const float* __restrict__ src, bf16_t* __restrict__ dst,
                        size_t n) {
  for (size_t i = (size_t)blockIdx.x * blockDim.x + threadIdx.x; i < n;
       i += (size_t)gridDim.x * blockDim.x)
    dst[i] = (bf16_t)src[i];
}

// ---------------------------------------------------------------------------
// Pack feature map into sequence-major bf16 [S][T][C] with scene-context
// fused at t==0 and t==T-1. mode 0: row scan (s=n*H+h, t=w);
// mode 1: col scan (s=n*W+w, t=h). Generic source strides (handles NCHW lf
// and NHWC intermediate maps).
// ---------------------------------------------------------------------------
__global__ void pack_seq(const float* __restrict__ src,
                         const float* __restrict__ scene,
                         bf16_t* __restrict__ xq,
                         int N, int Hh, int Ww, int C,
                         long sn, long sh, long sw, long scs, int mode) {
  size_t total = (size_t)N * Hh * Ww * C;
  int T = (mode == 0) ? Ww : Hh;
  for (size_t o = (size_t)blockIdx.x * blockDim.x + threadIdx.x; o < total;
       o += (size_t)gridDim.x * blockDim.x) {
    int c = (int)(o % C);
    size_t mt = o / C;
    int t = (int)(mt % T);
    size_t s = mt / T;
    int n, hh, wc;
    if (mode == 0) { n = (int)(s / Hh); hh = (int)(s % Hh); wc = t; }
    else           { n = (int)(s / Ww); wc = (int)(s % Ww); hh = t; }
    float v = src[(size_t)n * sn + (size_t)hh * sh + (size_t)wc * sw +
                  (size_t)c * scs];
    if (t == 0 || t == T - 1) v += scene[(size_t)n * C + c];
    xq[o] = (bf16_t)v;
  }
}

// ---------------------------------------------------------------------------
// Input projection GEMM: gx[m][n] = sum_k xq[m][k]*Wb[n][k] + bias[n],
// M = 38400, N = 1024 (dir0 | dir1), K in {512, 256}. Block tile 64x128,
// 8 waves, each wave = 1 M-frag x 4 N-frags. Tiles staged straight into LDS
// with global_load_async_to_lds_b128 (ASYNCcnt-tracked, no VGPR round-trip).
// ---------------------------------------------------------------------------
__global__ __launch_bounds__(256) void gemm_gx(
    const bf16_t* __restrict__ xq, const bf16_t* __restrict__ wb,
    const float* __restrict__ bias, bf16_t* __restrict__ gx, int K) {
  __shared__ bf16_t At[64][40];    // row pitch 80B = 5x16B -> 16B-aligned chunks
  __shared__ bf16_t Bt[128][40];
  const int tid = threadIdx.x;
  const int lane = tid & 31, w = tid >> 5;
  const int m0 = blockIdx.x * 64;
  const int n0 = blockIdx.y * 128;
  const int mi = w >> 1;         // 0..3
  const int njb = (w & 1) * 4;   // 0 or 4

  // Per-thread staging coordinates: A = 64 rows x 32 k (16B per thread),
  // B = 128 rows x 32 k (32B per thread, two b128 via offset:16).
  const int arow = tid >> 2;         // 0..63
  const int akk  = (tid & 3) * 8;    // bf16 elems: 0,8,16,24
  const int brow = tid >> 1;         // 0..127
  const unsigned a_lds = lds_off(&At[arow][akk]);
  const unsigned b_lds = lds_off(&Bt[brow][0]);

  v8f acc[4];
  for (int j = 0; j < 4; ++j)
    for (int r = 0; r < 8; ++r) acc[j][r] = 0.0f;

  const int kh = lane >> 4;
  for (int k0 = 0; k0 < K; k0 += 32) {
    __syncthreads();   // previous iteration's LDS readers done
    {
      unsigned a_go = (unsigned)(((size_t)(m0 + arow) * K + k0 + akk) *
                                 sizeof(bf16_t));
      unsigned b_go = (unsigned)(((size_t)(n0 + brow) * K + k0) *
                                 sizeof(bf16_t));
      asm volatile("global_load_async_to_lds_b128 %0, %1, %2"
                   :: "v"(a_lds), "v"(a_go), "s"(xq) : "memory");
      asm volatile("global_load_async_to_lds_b128 %0, %1, %2"
                   :: "v"(b_lds), "v"(b_go), "s"(wb) : "memory");
      asm volatile("global_load_async_to_lds_b128 %0, %1, %2 offset:16"
                   :: "v"(b_lds), "v"(b_go), "s"(wb) : "memory");
      asm volatile("s_wait_asynccnt 0x0" ::: "memory");
    }
    __syncthreads();   // tiles visible to all waves
    v16bf a;
    {
      int ar = mi * 16 + (lane & 15);
      for (int e = 0; e < 16; ++e)
        a[e] = At[ar][(kh << 4) + e];
    }
    for (int j = 0; j < 4; ++j) {
      v16bf b;
      int br = (njb + j) * 16 + (lane & 15);
      for (int e = 0; e < 16; ++e) b[e] = Bt[br][(kh << 4) + e];
      acc[j] = __builtin_amdgcn_wmma_f32_16x16x32_bf16(
          false, a, false, b, (short)0, acc[j], false, false);
    }
  }
  for (int j = 0; j < 4; ++j) {
    int n = n0 + (njb + j) * 16 + (lane & 15);
    float bv = bias[n];
    for (int r = 0; r < 8; ++r) {
      int m = m0 + mi * 16 + ((lane >> 4) << 3) + r;
      gx[(size_t)m * 1024 + n] = (bf16_t)(acc[j][r] + bv);
    }
  }
}

// ---------------------------------------------------------------------------
// Persistent bidirectional LSTM scan. One block = 16 sequences, full T loop.
// blockIdx.y = direction. Whh B-fragments preloaded into registers (16 frags
// per wave), h kept in LDS as the A operand, c in per-thread registers.
// Per step: g[16][512] = gx_tile + h@Whh^T via 4 K-chunk WMMAs per N-tile.
// The next step's gx tile is prefetched (global_prefetch_b8) behind the WMMAs.
// ---------------------------------------------------------------------------
__global__ __launch_bounds__(256) void lstm_scan(
    const bf16_t* __restrict__ gx, const bf16_t* __restrict__ whh_bf,
    float* __restrict__ y, int T, int mode, int Hh, int Ww) {
  __shared__ bf16_t hbuf[16][136];
  __shared__ float gbuf[16][520];
  const int tid = threadIdx.x;
  const int lane = tid & 31, w = tid >> 5;
  const int dir = blockIdx.y;
  const int s0 = blockIdx.x * 16;
  const int kh = lane >> 4;

  // Preload Whh fragments: wave w owns N-tiles (w*4 .. w*4+3) of 512 cols.
  v16bf bfrag[4][4];
  for (int i = 0; i < 4; ++i) {
    int n = (w * 4 + i) * 16 + (lane & 15);
    const bf16_t* wr = whh_bf + (size_t)(dir * 512 + n) * 128;
    for (int kc = 0; kc < 4; ++kc)
      for (int e = 0; e < 16; ++e)
        bfrag[i][kc][e] = wr[kc * 32 + (kh << 4) + e];
  }

  float creg[8];
  for (int j = 0; j < 8; ++j) creg[j] = 0.0f;
  for (int i = 0; i < 8; ++i) {  // zero h
    int idx = tid * 8 + i;
    hbuf[idx >> 7][idx & 127] = (bf16_t)0.0f;
  }
  __syncthreads();

  const int erow = tid >> 4;
  const int ecol0 = (tid & 15) * 8;

  for (int t = 0; t < T; ++t) {
    int tt = dir ? (T - 1 - t) : t;
    // A fragments from current h (one per K-chunk of 32)
    v16bf afr[4];
    {
      int ar = lane & 15;
      for (int kc = 0; kc < 4; ++kc)
        for (int e = 0; e < 16; ++e)
          afr[kc][e] = hbuf[ar][kc * 32 + (kh << 4) + e];
    }
    for (int i = 0; i < 4; ++i) {
      int nloc = (w * 4 + i) * 16 + (lane & 15);  // 0..511
      v8f c;
      for (int r = 0; r < 8; ++r) {
        int srow = s0 + ((lane >> 4) << 3) + r;
        c[r] = (float)gx[((size_t)srow * T + tt) * 1024 + dir * 512 + nloc];
      }
      if (t + 1 < T) {  // hide next step's gx latency behind the WMMAs
        int tn = dir ? (T - 2 - t) : (t + 1);
        int srow = s0 + ((lane >> 4) << 3);
        __builtin_prefetch(
            gx + ((size_t)srow * T + tn) * 1024 + dir * 512 + nloc, 0, 1);
      }
      for (int kc = 0; kc < 4; ++kc)
        c = __builtin_amdgcn_wmma_f32_16x16x32_bf16(
            false, afr[kc], false, bfrag[i][kc], (short)0, c, false, false);
      for (int r = 0; r < 8; ++r)
        gbuf[((lane >> 4) << 3) + r][nloc] = c[r];
    }
    __syncthreads();
    // Gate math (i,f,g,o each 128 wide), update c/h, emit h.
    for (int j = 0; j < 8; ++j) {
      int u = ecol0 + j;
      float iv = gbuf[erow][u];
      float fv = gbuf[erow][128 + u];
      float gv = gbuf[erow][256 + u];
      float ov = gbuf[erow][384 + u];
      float cc = sigmf(fv) * creg[j] + sigmf(iv) * tanhf(gv);
      creg[j] = cc;
      float hh = sigmf(ov) * tanhf(cc);
      hbuf[erow][u] = (bf16_t)hh;
      int s = s0 + erow;
      size_t oidx;
      if (mode == 0) {
        oidx = ((size_t)s * Ww + tt) * 256;             // s = n*Hh + h, tt = w
      } else {
        int n = s / Ww, wc = s % Ww;                    // s = n*Ww + w, tt = h
        oidx = (((size_t)n * Hh + tt) * Ww + wc) * 256;
      }
      y[oidx + dir * 128 + u] = hh;
    }
    __syncthreads();
  }
}

// ---------------------------------------------------------------------------
// 1x1 conv (256 -> 1) + sigmoid.
// ---------------------------------------------------------------------------
__global__ void conv_sig(const float* __restrict__ y,
                         const float* __restrict__ cw,
                         const float* __restrict__ cb,
                         float* __restrict__ score) {
  int p = blockIdx.x * blockDim.x + threadIdx.x;
  if (p >= 8 * 60 * 80) return;
  const float* row = y + (size_t)p * 256;
  float acc = cb[0];
  for (int c = 0; c < 256; ++c) acc += row[c] * cw[c];
  score[p] = 1.0f / (1.0f + __expf(-acc));
}

// ---------------------------------------------------------------------------
// Bilinear upsample, align_corners=True: [8,60,80] -> [8,480,640].
// ---------------------------------------------------------------------------
__global__ void upsample_ac(const float* __restrict__ score,
                            float* __restrict__ out) {
  const int OH = 480, OW = 640, IH = 60, IW = 80;
  int o = blockIdx.x * blockDim.x + threadIdx.x;
  if (o >= 8 * OH * OW) return;
  int n = o / (OH * OW);
  int rem = o % (OH * OW);
  int oy = rem / OW, ox = rem % OW;
  float sy = oy * ((float)(IH - 1) / (OH - 1));
  float sx = ox * ((float)(IW - 1) / (OW - 1));
  int y0 = (int)sy; if (y0 > IH - 2) y0 = IH - 2;
  int x0 = (int)sx; if (x0 > IW - 2) x0 = IW - 2;
  float ty = sy - y0, tx = sx - x0;
  const float* sp = score + (size_t)n * IH * IW;
  float a = sp[y0 * IW + x0],       b = sp[y0 * IW + x0 + 1];
  float c = sp[(y0 + 1) * IW + x0], d = sp[(y0 + 1) * IW + x0 + 1];
  float top = a + (b - a) * tx, bot = c + (d - c) * tx;
  out[o] = top + (bot - top) * ty;
}

// ---------------------------------------------------------------------------
// Host orchestration.
// ---------------------------------------------------------------------------
static inline size_t aln(size_t x) { return (x + 255) & ~(size_t)255; }

extern "C" void kernel_launch(void* const* d_in, const int* in_sizes, int n_in,
                              void* d_out, int out_size, void* d_ws, size_t ws_size,
                              hipStream_t stream) {
  (void)in_sizes; (void)n_in; (void)out_size; (void)ws_size;
  const float* lf      = (const float*)d_in[0];   // [8,512,60,80]
  const float* ctx     = (const float*)d_in[1];   // [8,128]
  const float* fc1_w   = (const float*)d_in[2];
  const float* fc1_b   = (const float*)d_in[3];
  const float* fcr_w   = (const float*)d_in[4];
  const float* fcr_b   = (const float*)d_in[5];
  const float* conv_w  = (const float*)d_in[18];
  const float* conv_b  = (const float*)d_in[19];

  const size_t Mseq = 38400;  // S*T for both scan orientations
  char* p = (char*)d_ws;
  float*  sc1    = (float*)p;  p += aln(8 * 512 * sizeof(float));
  float*  scr    = (float*)p;  p += aln(8 * 256 * sizeof(float));
  bf16_t* wih_bf = (bf16_t*)p; p += aln((size_t)1024 * 512 * sizeof(bf16_t));
  bf16_t* whh_bf = (bf16_t*)p; p += aln((size_t)1024 * 128 * sizeof(bf16_t));
  bf16_t* xq     = (bf16_t*)p; p += aln(Mseq * 512 * sizeof(bf16_t));
  bf16_t* gx     = (bf16_t*)p; p += aln(Mseq * 1024 * sizeof(bf16_t));
  float*  ybuf0  = (float*)p;  p += aln(Mseq * 256 * sizeof(float));
  float*  ybuf1  = (float*)p;  p += aln(Mseq * 256 * sizeof(float));
  float*  score  = (float*)p;  p += aln(Mseq * sizeof(float));

  fc_scene<<<24, 256, 0, stream>>>(ctx, fc1_w, fc1_b, fcr_w, fcr_b, sc1, scr);

  struct Layer {
    const float *wih, *whh, *b;
    int mode;      // 0 = row scan, 1 = col scan
    int C;         // input channels
    const float* src;
    long sn, sh, sw, scs;
    const float* scene;
    float* out;
  };
  Layer layers[4] = {
    { (const float*)d_in[6],  (const float*)d_in[7],  (const float*)d_in[8],
      0, 512, lf,    512L*60*80, 80, 1, 60L*80, sc1, ybuf0 },       // H1 (NCHW src)
    { (const float*)d_in[9],  (const float*)d_in[10], (const float*)d_in[11],
      1, 256, ybuf0, 60L*80*256, 80L*256, 256, 1, scr, ybuf1 },     // V1
    { (const float*)d_in[12], (const float*)d_in[13], (const float*)d_in[14],
      0, 256, ybuf1, 60L*80*256, 80L*256, 256, 1, scr, ybuf0 },     // H2
    { (const float*)d_in[15], (const float*)d_in[16], (const float*)d_in[17],
      1, 256, ybuf0, 60L*80*256, 80L*256, 256, 1, scr, ybuf1 },     // V2
  };

  for (int L = 0; L < 4; ++L) {
    const Layer& ly = layers[L];
    const int K = ly.C;
    const int S = (ly.mode == 0) ? 480 : 640;
    const int T = (ly.mode == 0) ? 80 : 60;

    to_bf16<<<2048, 256, 0, stream>>>(ly.wih, wih_bf, (size_t)1024 * K);
    to_bf16<<<512, 256, 0, stream>>>(ly.whh, whh_bf, (size_t)1024 * 128);
    pack_seq<<<4096, 256, 0, stream>>>(ly.src, ly.scene, xq, 8, 60, 80, ly.C,
                                       ly.sn, ly.sh, ly.sw, ly.scs, ly.mode);
    gemm_gx<<<dim3(600, 8), 256, 0, stream>>>(xq, wih_bf, ly.b, gx, K);
    lstm_scan<<<dim3(S / 16, 2), 256, 0, stream>>>(gx, whh_bf, ly.out, T,
                                                   ly.mode, 60, 80);
  }

  conv_sig<<<150, 256, 0, stream>>>(ybuf1, conv_w, conv_b, score);
  upsample_ac<<<(8 * 480 * 640 + 255) / 256, 256, 0, stream>>>(score,
                                                               (float*)d_out);
}
